// EfficientDETRDecoder_29970281792174
// MI455X (gfx1250) — matrix operand order repaired and checked by reading, not compiled, not run on hardware
//
#include <hip/hip_runtime.h>
#include <hip/hip_bf16.h>
#include <math.h>

typedef __attribute__((ext_vector_type(16))) _Float16 v16h;
typedef __attribute__((ext_vector_type(8)))  float    v8f;

#define C_DIM 256
#define NQ 300
#define HEADS 8
#define POINTS 4
#define N_LAYERS 2
#define BATCH 8
#define HH 64
#define WW 64
#define HW_SZ 4096
#define MQ (BATCH*NQ)     /* 2400  */
#define MV (BATCH*HW_SZ)  /* 32768 */

// ---------------------------------------------------------------------------
// WMMA fragment address helpers (CDNA5 ISA 7.12.2, wave32)
// A operand: 16x32 (MxK) f16, per-lane v16h.
//   lanes 0-15: M=lane, halves 0-7 -> K 0-7, halves 8-15 -> K 16-23
//   lanes16-31: M=lane-16, halves 0-7 -> K 8-15, halves 8-15 -> K 24-31
// B operand: 32x16 (KxN) f16, per-lane v16h.
//   lanes 0-15: N=lane, halves 0-7 -> K 0-7, 8-15 -> K 16-23;  lanes 16-31: N=lane-16, 0-7 -> K 8-15, 8-15 -> K 24-31
// C/D: 8 VGPRs f32: lane<16 -> N=lane, M=r; lane>=16 -> N=lane-16, M=r+8
// ---------------------------------------------------------------------------
__device__ __forceinline__ size_t afrag_off(int m, int k, int Kdim) {
  int tile   = (m >> 4) * (Kdim >> 5) + (k >> 5);
  int kk     = k & 31;
  int laneHi = (kk >> 3) & 1;
  int h      = (kk & 7) + (((kk >> 4) & 1) << 3);
  int lane   = (m & 15) + (laneHi << 4);
  return (size_t)tile * 512 + lane * 16 + h;
}
__device__ __forceinline__ size_t bfrag_off(int k, int n, int Ndim) {
  int tile = (k >> 5) * (Ndim >> 4) + (n >> 4);
  int lane = (n & 15) + (((k >> 3) & 1) << 4);
  int h    = (k & 7) + (((k >> 4) & 1) << 3);
  return (size_t)tile * 512 + lane * 16 + h;
}

// ---------------------------------------------------------------------------
// Generic WMMA GEMM: out = act(Afrag[M,K] @ Bfrag[K,N] + bias) (+ resid)
// One wave per 16x(16*NT) output strip; A fragment reused across NT wmmas.
// ---------------------------------------------------------------------------
template <int ACT, int NT>
__global__ void gemm_kernel(const _Float16* __restrict__ Afrag,
                            const _Float16* __restrict__ Bfrag,
                            const float* __restrict__ bias,
                            const float* __restrict__ resid,
                            float* __restrict__ oF32,
                            _Float16* __restrict__ oFrag,
                            _Float16* __restrict__ oHalf,
                            int M, int K, int N) {
  const int mtiles = M >> 4;
  int mtile = blockIdx.x * (blockDim.x >> 5) + (threadIdx.x >> 5);
  if (mtile >= mtiles) return;           // whole-wave uniform: EXEC stays all-1s
  const int lane   = threadIdx.x & 31;
  const int nt0    = blockIdx.y * NT;
  const int ktiles = K >> 5;
  const int ntTot  = N >> 4;

  v8f acc[NT];
#pragma unroll
  for (int j = 0; j < NT; ++j)
#pragma unroll
    for (int r = 0; r < 8; ++r) acc[j][r] = 0.0f;

  const _Float16* Ab = Afrag + (size_t)mtile * ktiles * 512 + lane * 16;
  for (int kt = 0; kt < ktiles; ++kt) {
    v16h a = *(const v16h*)(Ab + (size_t)kt * 512);
#pragma unroll
    for (int j = 0; j < NT; ++j) {
      v16h b = *(const v16h*)(Bfrag + ((size_t)kt * ntTot + nt0 + j) * 512 + lane * 16);
      acc[j] = __builtin_amdgcn_wmma_f32_16x16x32_f16(
          false, a, false, b, (short)0, acc[j], false, false);
    }
  }

  const int rowBase = mtile * 16 + ((lane >> 4) << 3);
  const int colLo   = lane & 15;
#pragma unroll
  for (int j = 0; j < NT; ++j) {
    int n    = (nt0 + j) * 16 + colLo;
    float bv = bias ? bias[n] : 0.0f;
#pragma unroll
    for (int r = 0; r < 8; ++r) {
      int m   = rowBase + r;
      float v = acc[j][r] + bv;
      if (ACT == 1) v = fmaxf(v, 0.0f);
      if (ACT == 2) v = 1.0f / (1.0f + expf(-v));
      if (resid) v += resid[(size_t)m * N + n];
      if (oF32)  oF32[(size_t)m * N + n] = v;
      if (oHalf) oHalf[(size_t)m * N + n] = (_Float16)v;
      if (oFrag) oFrag[afrag_off(m, n, N)] = (_Float16)v;
    }
  }
}

// f32 weight [K,Nsrc] row-major -> f16 B-fragment, N padded to Npad with zeros
__global__ void wconv_kernel(const float* __restrict__ W, _Float16* __restrict__ Bf,
                             int K, int Nsrc, int Npad) {
  int idx = blockIdx.x * blockDim.x + threadIdx.x;
  if (idx >= K * Npad) return;
  int k = idx / Npad, n = idx % Npad;
  float v = (n < Nsrc) ? W[(size_t)k * Nsrc + n] : 0.0f;
  Bf[bfrag_off(k, n, Npad)] = (_Float16)v;
}

// f32 [M,K] row-major -> f16 A-fragment
__global__ void tofrag_kernel(const float* __restrict__ X, _Float16* __restrict__ F,
                              int M, int K) {
  int idx = blockIdx.x * blockDim.x + threadIdx.x;
  if (idx >= M * K) return;
  int m = idx / K, k = idx % K;
  F[afrag_off(m, k, K)] = (_Float16)X[idx];
}

__global__ void biaspad_kernel(const float* __restrict__ src, float* __restrict__ dst, int n) {
  int i = threadIdx.x;
  if (i < 16) dst[i] = (i < n) ? src[i] : 0.0f;
}

// feat_map[B,C,H,W] + sine positional encoding -> f16 A-fragment [MV, 256]
__global__ void featpos_kernel(const float* __restrict__ feat, _Float16* __restrict__ F) {
  int c  = threadIdx.x;          // 0..255
  int hw = blockIdx.x;           // 0..4095
  int b  = blockIdx.y;           // 0..7
  int ci = c & 127;
  int j  = ci >> 1;
  float dt    = powf(10000.0f, (float)j / 64.0f);
  float coord = (c < 128 ? (float)(hw >> 6) : (float)(hw & 63)) * (1.0f / 64.0f);
  float v     = coord / dt;
  float pos   = (ci & 1) ? cosf(v) : sinf(v);
  float x = feat[((size_t)b * C_DIM + c) * HW_SZ + hw] + pos;
  F[afrag_off(b * HW_SZ + hw, c, C_DIM)] = (_Float16)x;
}

// q init: broadcast query_embed over batch
__global__ void initq_kernel(const float* __restrict__ qe, float* __restrict__ q) {
  int idx = blockIdx.x * blockDim.x + threadIdx.x;
  if (idx >= MQ * C_DIM) return;
  int m = idx >> 8, c = idx & 255;
  q[idx] = qe[(m % NQ) * C_DIM + c];
}

// LayerNorm over C=256, emits swizzled f16 fragment
__global__ void ln_kernel(const float* __restrict__ X, const float* __restrict__ g,
                          const float* __restrict__ b, _Float16* __restrict__ F) {
  int m = blockIdx.x, c = threadIdx.x;
  __shared__ float sb[256];
  float x = X[(size_t)m * C_DIM + c];
  sb[c] = x; __syncthreads();
  for (int st = 128; st > 0; st >>= 1) { if (c < st) sb[c] += sb[c + st]; __syncthreads(); }
  float mean = sb[0] * (1.0f / 256.0f);
  __syncthreads();
  float d = x - mean;
  sb[c] = d * d; __syncthreads();
  for (int st = 128; st > 0; st >>= 1) { if (c < st) sb[c] += sb[c + st]; __syncthreads(); }
  float r = rsqrtf(sb[0] * (1.0f / 256.0f) + 1e-5f);
  F[afrag_off(m, c, C_DIM)] = (_Float16)(d * r * g[c] + b[c]);
}

// MHA core: one wave per (b,h,query); lane owns one d of 32; online softmax
__global__ void attn_kernel(const float* __restrict__ Q, const float* __restrict__ Kp,
                            const float* __restrict__ Vp, float* __restrict__ O) {
  int wid = blockIdx.x * (blockDim.x >> 5) + (threadIdx.x >> 5);
  if (wid >= MQ * HEADS) return;
  int lane = threadIdx.x & 31;
  int q = wid % NQ;
  int h = (wid / NQ) & (HEADS - 1);
  int b = wid / (NQ * HEADS);
  size_t qoff = ((size_t)(b * NQ + q)) * C_DIM + h * 32 + lane;
  size_t base = ((size_t)b * NQ) * C_DIM + h * 32 + lane;
  float qv = Q[qoff] * 0.1767766952966369f;   // 1/sqrt(32)
  float mrun = -3.0e38f, lrun = 0.0f, oacc = 0.0f;
  for (int k = 0; k < NQ; ++k) {
    float p = qv * Kp[base + (size_t)k * C_DIM];
    for (int o = 16; o > 0; o >>= 1) p += __shfl_xor(p, o, 32);
    float mn   = fmaxf(mrun, p);
    float corr = expf(mrun - mn);
    float e    = expf(p - mn);
    lrun = lrun * corr + e;
    oacc = oacc * corr + e * Vp[base + (size_t)k * C_DIM];
    mrun = mn;
  }
  O[qoff] = oacc / lrun;
}

// Deformable attention sampling: thread = (m=b*300+q, h, dd); P=4 bilinear taps
__global__ void deform_kernel(const float* __restrict__ refb, const float* __restrict__ Off,
                              const float* __restrict__ Aw, const _Float16* __restrict__ vhalf,
                              float* __restrict__ Dout) {
  int m = blockIdx.x;
  int t = threadIdx.x;
  int dd = t & 31, h = t >> 5;
  int b = m / NQ;
  float rx = refb[m * 16 + 0], ry = refb[m * 16 + 1];
  // softmax over 4 attention weights (uniform across the 32 dd lanes)
  float a[POINTS]; float amax = -3.0e38f;
  for (int p = 0; p < POINTS; ++p) { a[p] = Aw[m * 32 + h * POINTS + p]; amax = fmaxf(amax, a[p]); }
  float asum = 0.0f;
  for (int p = 0; p < POINTS; ++p) { a[p] = expf(a[p] - amax); asum += a[p]; }
  float inv = 1.0f / asum;
  float acc = 0.0f;
  for (int p = 0; p < POINTS; ++p) {
    float ox = Off[m * 64 + (h * POINTS + p) * 2 + 0] * (0.1f / 64.0f);
    float oy = Off[m * 64 + (h * POINTS + p) * 2 + 1] * (0.1f / 64.0f);
    float lx = fminf(fmaxf(rx + ox, 0.0f), 1.0f);
    float ly = fminf(fmaxf(ry + oy, 0.0f), 1.0f);
    float x = lx * 64.0f - 0.5f, y = ly * 64.0f - 0.5f;
    float x0f = floorf(x), y0f = floorf(y);
    int x0 = (int)x0f, y0 = (int)y0f;
    float wx1 = x - x0f, wx0 = 1.0f - wx1;
    float wy1 = y - y0f, wy0 = 1.0f - wy1;
    float s = 0.0f;
    int ch = h * 32 + dd;
#pragma unroll
    for (int cy = 0; cy < 2; ++cy)
#pragma unroll
      for (int cx = 0; cx < 2; ++cx) {
        int xi = x0 + cx, yi = y0 + cy;
        if (xi >= 0 && xi < WW && yi >= 0 && yi < HH) {
          float w = (cx ? wx1 : wx0) * (cy ? wy1 : wy0);
          s += w * (float)vhalf[((size_t)(b * HW_SZ + yi * WW + xi)) * C_DIM + ch];
        }
      }
    acc += a[p] * inv * s;
  }
  Dout[(size_t)m * C_DIM + h * 32 + dd] = acc;
}

__global__ void finalize_kernel(const float* __restrict__ BBox, const float* __restrict__ Obj,
                                float* __restrict__ out, float* __restrict__ refbuf,
                                int li, int update_ref) {
  int m = blockIdx.x * blockDim.x + threadIdx.x;
  if (m >= MQ) return;
  float b0 = BBox[m * 16 + 0], b1 = BBox[m * 16 + 1];
  float b2 = BBox[m * 16 + 2], b3 = BBox[m * 16 + 3];
  float* o = out + ((size_t)li * MQ + m) * 5;
  o[0] = b0; o[1] = b1; o[2] = b2; o[3] = b3; o[4] = Obj[m * 16];
  if (update_ref) {
    refbuf[m * 16 + 0] = (b0 + b2) * 0.5f;
    refbuf[m * 16 + 1] = (b1 + b3) * 0.5f;
  }
}

// ---------------------------------------------------------------------------
// Host side
// ---------------------------------------------------------------------------
static void gemm(hipStream_t s, const _Float16* A, const _Float16* B, const float* bias,
                 const float* resid, float* oF32, _Float16* oFrag, _Float16* oHalf,
                 int M, int K, int N, int act) {
  int NT = (N % 64 == 0) ? 4 : ((N % 32 == 0) ? 2 : 1);
  dim3 blk(128, 1, 1);
  dim3 grd(((M >> 4) + 3) / 4, N / (16 * NT), 1);
#define GL(A_, NT_) gemm_kernel<A_, NT_><<<grd, blk, 0, s>>>(A, B, bias, resid, oF32, oFrag, oHalf, M, K, N)
  switch (act * 8 + NT) {
    case 0 * 8 + 1: GL(0, 1); break;
    case 0 * 8 + 2: GL(0, 2); break;
    case 0 * 8 + 4: GL(0, 4); break;
    case 1 * 8 + 1: GL(1, 1); break;
    case 1 * 8 + 2: GL(1, 2); break;
    case 1 * 8 + 4: GL(1, 4); break;
    case 2 * 8 + 1: GL(2, 1); break;
    case 2 * 8 + 2: GL(2, 2); break;
    case 2 * 8 + 4: GL(2, 4); break;
  }
#undef GL
}

extern "C" void kernel_launch(void* const* d_in, const int* in_sizes, int n_in,
                              void* d_out, int out_size, void* d_ws, size_t ws_size,
                              hipStream_t stream) {
  (void)in_sizes; (void)n_in; (void)out_size; (void)ws_size;
  // Input order: jax pytree (sorted dict keys): feat_map, layers[0..1] (each: bbox.b1,
  // bbox.b2, bbox.w1, bbox.w2, ca.aw_b, ca.aw_w, ca.off_b, ca.off_w, ca.out_b, ca.out_w,
  // ca.val_b, ca.val_w, ffn.b1, ffn.b2, ffn.w1, ffn.w2, ln1_b, ln1_g, ln2_b, ln2_g,
  // ln3_b, ln3_g, obj.b, obj.w, sa.bk, sa.bo, sa.bq, sa.bv, sa.wk, sa.wo, sa.wq, sa.wv),
  // then query_embed, ref_b, ref_w.
  const float* feat_map = (const float*)d_in[0];
  const float* qe    = (const float*)d_in[65];
  const float* ref_b = (const float*)d_in[66];
  const float* ref_w = (const float*)d_in[67];
  struct LayerIn {
    const float *bbox_b1, *bbox_b2, *bbox_w1, *bbox_w2;
    const float *aw_b, *aw_w, *off_b, *off_w, *out_b, *out_w, *val_b, *val_w;
    const float *ffn_b1, *ffn_b2, *ffn_w1, *ffn_w2;
    const float *ln1_b, *ln1_g, *ln2_b, *ln2_g, *ln3_b, *ln3_g;
    const float *obj_b, *obj_w;
    const float *bk, *bo, *bq, *bv, *wk, *wo, *wq, *wv;
  } L[N_LAYERS];
  for (int li = 0; li < N_LAYERS; ++li) {
    const float** p = (const float**)&L[li];
    for (int j = 0; j < 32; ++j) p[j] = (const float*)d_in[1 + 32 * li + j];
  }
  float* out = (float*)d_out;

  char* wsp = (char*)d_ws;
  auto alloc = [&](size_t bytes) {
    char* p = wsp;
    wsp += (bytes + 255) & ~(size_t)255;
    return p;
  };
  _Float16* featfrag = (_Float16*)alloc((size_t)MV * C_DIM * 2);
  _Float16* vhalf    = (_Float16*)alloc((size_t)MV * C_DIM * 2);
  float*    qbuf   = (float*)alloc((size_t)MQ * C_DIM * 4);
  _Float16* qfrag  = (_Float16*)alloc((size_t)MQ * C_DIM * 2);
  _Float16* qnfrag = (_Float16*)alloc((size_t)MQ * C_DIM * 2);
  _Float16* tfrag  = (_Float16*)alloc((size_t)MQ * C_DIM * 2);
  _Float16* h1frag = (_Float16*)alloc((size_t)MQ * 512 * 2);
  float* Qp  = (float*)alloc((size_t)MQ * C_DIM * 4);
  float* Kp  = (float*)alloc((size_t)MQ * C_DIM * 4);
  float* Vp  = (float*)alloc((size_t)MQ * C_DIM * 4);
  float* Tmp = (float*)alloc((size_t)MQ * C_DIM * 4);
  float* Off = (float*)alloc((size_t)MQ * 64 * 4);
  float* Aw  = (float*)alloc((size_t)MQ * 32 * 4);
  float* refbuf = (float*)alloc((size_t)MQ * 16 * 4);
  float* BBox   = (float*)alloc((size_t)MQ * 16 * 4);
  float* Obj    = (float*)alloc((size_t)MQ * 16 * 4);
  struct LayerW {
    _Float16 *wq, *wk, *wv, *wo, *offw, *aww, *valw, *outw, *w1, *w2, *bw1, *bw2, *objw;
    float *bb2pad, *objbpad;
  } W[N_LAYERS];
  for (int li = 0; li < N_LAYERS; ++li) {
    W[li].wq   = (_Float16*)alloc(256 * 256 * 2);
    W[li].wk   = (_Float16*)alloc(256 * 256 * 2);
    W[li].wv   = (_Float16*)alloc(256 * 256 * 2);
    W[li].wo   = (_Float16*)alloc(256 * 256 * 2);
    W[li].offw = (_Float16*)alloc(256 * 64 * 2);
    W[li].aww  = (_Float16*)alloc(256 * 32 * 2);
    W[li].valw = (_Float16*)alloc(256 * 256 * 2);
    W[li].outw = (_Float16*)alloc(256 * 256 * 2);
    W[li].w1   = (_Float16*)alloc(256 * 512 * 2);
    W[li].w2   = (_Float16*)alloc(512 * 256 * 2);
    W[li].bw1  = (_Float16*)alloc(256 * 256 * 2);
    W[li].bw2  = (_Float16*)alloc(256 * 16 * 2);
    W[li].objw = (_Float16*)alloc(256 * 16 * 2);
    W[li].bb2pad  = (float*)alloc(64);
    W[li].objbpad = (float*)alloc(64);
  }
  _Float16* refwfrag = (_Float16*)alloc(256 * 16 * 2);
  float* refbpad = (float*)alloc(64);

  auto wc = [&](const float* src, _Float16* dst, int K, int Nsrc, int Npad) {
    int n = K * Npad;
    wconv_kernel<<<(n + 255) / 256, 256, 0, stream>>>(src, dst, K, Nsrc, Npad);
  };
  auto tf = [&](const float* src, _Float16* dst, int M, int K) {
    int n = M * K;
    tofrag_kernel<<<(n + 255) / 256, 256, 0, stream>>>(src, dst, M, K);
  };

  // --- weight conversion (every call; deterministic) ---
  for (int li = 0; li < N_LAYERS; ++li) {
    wc(L[li].wq, W[li].wq, 256, 256, 256);
    wc(L[li].wk, W[li].wk, 256, 256, 256);
    wc(L[li].wv, W[li].wv, 256, 256, 256);
    wc(L[li].wo, W[li].wo, 256, 256, 256);
    wc(L[li].off_w, W[li].offw, 256, 64, 64);
    wc(L[li].aw_w,  W[li].aww,  256, 32, 32);
    wc(L[li].val_w, W[li].valw, 256, 256, 256);
    wc(L[li].out_w, W[li].outw, 256, 256, 256);
    wc(L[li].ffn_w1, W[li].w1, 256, 512, 512);
    wc(L[li].ffn_w2, W[li].w2, 512, 256, 256);
    wc(L[li].bbox_w1, W[li].bw1, 256, 256, 256);
    wc(L[li].bbox_w2, W[li].bw2, 256, 4, 16);
    wc(L[li].obj_w,   W[li].objw, 256, 1, 16);
    biaspad_kernel<<<1, 16, 0, stream>>>(L[li].bbox_b2, W[li].bb2pad, 4);
    biaspad_kernel<<<1, 16, 0, stream>>>(L[li].obj_b,   W[li].objbpad, 1);
  }
  wc(ref_w, refwfrag, 256, 2, 16);
  biaspad_kernel<<<1, 16, 0, stream>>>(ref_b, refbpad, 2);

  // --- feature + positional encoding (once), q init, reference points ---
  featpos_kernel<<<dim3(HW_SZ, BATCH), 256, 0, stream>>>(feat_map, featfrag);
  initq_kernel<<<(MQ * C_DIM + 255) / 256, 256, 0, stream>>>(qe, qbuf);
  tf(qbuf, qfrag, MQ, C_DIM);
  gemm(stream, qfrag, refwfrag, refbpad, nullptr, refbuf, nullptr, nullptr, MQ, 256, 16, 2);

  for (int li = 0; li < N_LAYERS; ++li) {
    // -- self attention --
    ln_kernel<<<MQ, 256, 0, stream>>>(qbuf, L[li].ln1_g, L[li].ln1_b, qnfrag);
    gemm(stream, qnfrag, W[li].wq, L[li].bq, nullptr, Qp, nullptr, nullptr, MQ, 256, 256, 0);
    gemm(stream, qnfrag, W[li].wk, L[li].bk, nullptr, Kp, nullptr, nullptr, MQ, 256, 256, 0);
    gemm(stream, qnfrag, W[li].wv, L[li].bv, nullptr, Vp, nullptr, nullptr, MQ, 256, 256, 0);
    attn_kernel<<<MQ * HEADS / 8, 256, 0, stream>>>(Qp, Kp, Vp, Tmp);
    tf(Tmp, tfrag, MQ, C_DIM);
    gemm(stream, tfrag, W[li].wo, L[li].bo, qbuf, qbuf, nullptr, nullptr, MQ, 256, 256, 0);

    // -- deformable cross attention --
    ln_kernel<<<MQ, 256, 0, stream>>>(qbuf, L[li].ln2_g, L[li].ln2_b, qnfrag);
    gemm(stream, qnfrag, W[li].offw, L[li].off_b, nullptr, Off, nullptr, nullptr, MQ, 256, 64, 0);
    gemm(stream, qnfrag, W[li].aww,  L[li].aw_b,  nullptr, Aw,  nullptr, nullptr, MQ, 256, 32, 0);
    gemm(stream, featfrag, W[li].valw, L[li].val_b, nullptr, nullptr, nullptr, vhalf, MV, 256, 256, 0);
    deform_kernel<<<MQ, 256, 0, stream>>>(refbuf, Off, Aw, vhalf, Tmp);
    tf(Tmp, tfrag, MQ, C_DIM);
    gemm(stream, tfrag, W[li].outw, L[li].out_b, qbuf, qbuf, nullptr, nullptr, MQ, 256, 256, 0);

    // -- FFN --
    ln_kernel<<<MQ, 256, 0, stream>>>(qbuf, L[li].ln3_g, L[li].ln3_b, qnfrag);
    gemm(stream, qnfrag, W[li].w1, L[li].ffn_b1, nullptr, nullptr, h1frag, nullptr, MQ, 256, 512, 1);
    gemm(stream, h1frag, W[li].w2, L[li].ffn_b2, qbuf, qbuf, nullptr, nullptr, MQ, 512, 256, 0);

    // -- heads --
    tf(qbuf, qfrag, MQ, C_DIM);
    gemm(stream, qfrag, W[li].bw1, L[li].bbox_b1, nullptr, nullptr, tfrag, nullptr, MQ, 256, 256, 1);
    gemm(stream, tfrag, W[li].bw2, W[li].bb2pad, nullptr, BBox, nullptr, nullptr, MQ, 256, 16, 2);
    gemm(stream, qfrag, W[li].objw, W[li].objbpad, nullptr, Obj, nullptr, nullptr, MQ, 256, 16, 0);
    finalize_kernel<<<(MQ + 255) / 256, 256, 0, stream>>>(BBox, Obj, out, refbuf, li,
                                                          li == 0 ? 1 : 0);
  }
}
